// CausalSelfAttention_17197049053828
// MI455X (gfx1250) — compile-verified, matrix operand order checked by reading, not run on hardware
//
#include <hip/hip_runtime.h>
#include <hip/hip_bf16.h>

// ---------------------------------------------------------------------------
// Causal self-attention for MI455X (gfx1250), bf16 WMMA path + TDM staging.
//   B=4, T=2048, D=1024, H=16, DK=64
// Pipeline:
//   1) qkv_gemm : q/k/v = x @ W^T + b  (f32 in -> bf16 [B,H,T,DK]), scale on Q
//   2) attn     : flash attention, K tile staged via tensor_load_to_lds (TDM)
//   3) out_gemm : out = y @ Wo^T + bo  (A tile staged via TDM)
// ---------------------------------------------------------------------------

typedef __attribute__((ext_vector_type(16))) __bf16 v16bf;
typedef __attribute__((ext_vector_type(8)))  __bf16 v8bf;
typedef __attribute__((ext_vector_type(8)))  float  v8f;
typedef __attribute__((ext_vector_type(4)))  unsigned int u32x4;
typedef __attribute__((ext_vector_type(8)))  int i32x8;
typedef __attribute__((ext_vector_type(4)))  int i32x4;

#define WMMA_BF16(a, b, c) \
  __builtin_amdgcn_wmma_f32_16x16x32_bf16(false, (a), false, (b), (short)0, (c), false, false)

static __device__ __forceinline__ __bf16 f2bf(float f) { return (__bf16)f; }

static __device__ __forceinline__ v8f zero8() {
  v8f z;
#pragma unroll
  for (int i = 0; i < 8; ++i) z[i] = 0.0f;
  return z;
}

// Load one 16x32 A-frag (or mirrored B-frag) row slice: 8 bf16 at +k0, 8 at +k0+16.
static __device__ __forceinline__ v16bf load_frag(const __bf16* row, int k0) {
  v8bf lo = *(const v8bf*)(row + k0);
  v8bf hi = *(const v8bf*)(row + k0 + 16);
  return __builtin_shufflevector(lo, hi, 0,1,2,3,4,5,6,7,8,9,10,11,12,13,14,15);
}

// ---------------------------------------------------------------------------
// Tensor Data Mover: 2D bf16 tile load global->LDS with optional LDS row pad.
// D# built per CDNA5 ISA ch.8 (group0: count/lds/global/type, group1: dims).
// tile_w/row_stride in bf16 elements; pad codes per ISA (interval: 0=2 dw..;
// amount: 0=1 dw..). Issued by one wave; EXEC ignored by TDM.
// ---------------------------------------------------------------------------
static __device__ __forceinline__ void tdm_load_2d_bf16(
    unsigned int lds_off, const void* gptr,
    unsigned int tile_w, unsigned int tile_h, unsigned int row_stride,
    unsigned int pad_interval_code, unsigned int pad_amount_code) {
  unsigned long long ga = (unsigned long long)(size_t)gptr;
  u32x4 g0;
  g0[0] = 1u;                                            // count=1 (valid), user mode
  g0[1] = lds_off;                                       // lds_addr (bytes)
  g0[2] = (unsigned int)(ga & 0xFFFFFFFFu);              // global_addr[31:0]
  g0[3] = (unsigned int)((ga >> 32) & 0x01FFFFFFu)       // global_addr[56:32]
          | (2u << 30);                                  // type = 2 ("image")
  i32x8 g1;
  g1[0] = (int)((1u << 16)                               // data_size = 2 bytes
                | (1u << 20)                             // pad_enable
                | (pad_interval_code << 22)
                | (pad_amount_code << 25));
  g1[1] = (int)((tile_w & 0xFFFFu) << 16);               // tensor_dim0[15:0]
  g1[2] = (int)(((tile_w >> 16) & 0xFFFFu)               // tensor_dim0[31:16]
                | ((tile_h & 0xFFFFu) << 16));           // tensor_dim1[15:0]
  g1[3] = (int)(((tile_h >> 16) & 0xFFFFu)               // tensor_dim1[31:16]
                | ((tile_w & 0xFFFFu) << 16));           // tile_dim0
  g1[4] = (int)(tile_h & 0xFFFFu);                       // tile_dim1 (tile_dim2=0)
  g1[5] = (int)row_stride;                               // tensor_dim0_stride[31:0]
  g1[6] = 0;                                             // stride hi / dim1_stride lo
  g1[7] = 0;
  i32x4 g2 = (i32x4)0, g3 = (i32x4)0;                    // 2D: groups 2/3 unused
#if defined(__clang_major__) && __clang_major__ >= 23
  i32x8 g4 = (i32x8)0;
  __builtin_amdgcn_tensor_load_to_lds(g0, g1, g2, g3, g4, 0);
#else
  __builtin_amdgcn_tensor_load_to_lds(g0, g1, g2, g3, 0);
#endif
}

enum { Bn = 4, T = 2048, D = 1024, H = 16, DK = 64, M_TOT = Bn * T };
enum { LDS_STRIDE = 48 };  // 32 K-elems + 16 pad; 96B rows keep 16B alignment

// ---------------------------------------------------------------------------
// Kernel 1: fused QKV projection. blockIdx.z selects q/k/v.
// Tile: 128x128, Kstep 32, 256 threads (8 waves, each 32x64).
// ---------------------------------------------------------------------------
__global__ __launch_bounds__(256) void qkv_gemm(
    const float* __restrict__ x,
    const float* __restrict__ Wq, const float* __restrict__ bq,
    const float* __restrict__ Wk, const float* __restrict__ bk,
    const float* __restrict__ Wv, const float* __restrict__ bv,
    unsigned short* __restrict__ Qu, unsigned short* __restrict__ Ku,
    unsigned short* __restrict__ Vu) {
  const int z = blockIdx.z;
  const float* W    = (z == 0) ? Wq : (z == 1) ? Wk : Wv;
  const float* bias = (z == 0) ? bq : (z == 1) ? bk : bv;
  __bf16* dst = (__bf16*)((z == 0) ? Qu : (z == 1) ? Ku : Vu);
  const float scale = (z == 0) ? 0.125f : 1.0f;   // 1/sqrt(DK) folded into Q

  const int m0 = blockIdx.x * 128;
  const int n0 = blockIdx.y * 128;
  const int tid  = threadIdx.x;
  const int lane = tid & 31, ln = lane & 15, half = lane >> 4, k0 = half * 8;
  const int wave = tid >> 5, wm = wave >> 1, wn = wave & 1;

  __shared__ __align__(16) __bf16 As[128][LDS_STRIDE];
  __shared__ __align__(16) __bf16 Bs[128][LDS_STRIDE];

  v8f acc[2][4];
#pragma unroll
  for (int i = 0; i < 2; ++i)
#pragma unroll
    for (int j = 0; j < 4; ++j) acc[i][j] = zero8();

  const int srow = tid >> 1, scb = (tid & 1) * 16;

  for (int kk = 0; kk < D; kk += 32) {
    __syncthreads();
    // --- stage A (x, f32 -> bf16) and B (W, f32 -> bf16) ---
    {
      const float4* ap = (const float4*)(x + (size_t)(m0 + srow) * D + kk + scb);
      const float4* bp = (const float4*)(W + (size_t)(n0 + srow) * D + kk + scb);
      if (kk + 32 < D) {
        __builtin_prefetch(x + (size_t)(m0 + srow) * D + kk + 32 + scb, 0, 1);
        __builtin_prefetch(W + (size_t)(n0 + srow) * D + kk + 32 + scb, 0, 1);
      }
      float4 a0 = ap[0], a1 = ap[1], a2 = ap[2], a3 = ap[3];
      float4 b0 = bp[0], b1 = bp[1], b2 = bp[2], b3 = bp[3];
      v8bf pa0, pa1, pb0, pb1;
      pa0[0]=f2bf(a0.x); pa0[1]=f2bf(a0.y); pa0[2]=f2bf(a0.z); pa0[3]=f2bf(a0.w);
      pa0[4]=f2bf(a1.x); pa0[5]=f2bf(a1.y); pa0[6]=f2bf(a1.z); pa0[7]=f2bf(a1.w);
      pa1[0]=f2bf(a2.x); pa1[1]=f2bf(a2.y); pa1[2]=f2bf(a2.z); pa1[3]=f2bf(a2.w);
      pa1[4]=f2bf(a3.x); pa1[5]=f2bf(a3.y); pa1[6]=f2bf(a3.z); pa1[7]=f2bf(a3.w);
      pb0[0]=f2bf(b0.x); pb0[1]=f2bf(b0.y); pb0[2]=f2bf(b0.z); pb0[3]=f2bf(b0.w);
      pb0[4]=f2bf(b1.x); pb0[5]=f2bf(b1.y); pb0[6]=f2bf(b1.z); pb0[7]=f2bf(b1.w);
      pb1[0]=f2bf(b2.x); pb1[1]=f2bf(b2.y); pb1[2]=f2bf(b2.z); pb1[3]=f2bf(b2.w);
      pb1[4]=f2bf(b3.x); pb1[5]=f2bf(b3.y); pb1[6]=f2bf(b3.z); pb1[7]=f2bf(b3.w);
      *(v8bf*)&As[srow][scb]     = pa0;
      *(v8bf*)&As[srow][scb + 8] = pa1;
      *(v8bf*)&Bs[srow][scb]     = pb0;
      *(v8bf*)&Bs[srow][scb + 8] = pb1;
    }
    __syncthreads();
    // --- compute: 2 M-tiles x 4 N-tiles per wave ---
    v16bf af[2];
#pragma unroll
    for (int mt = 0; mt < 2; ++mt)
      af[mt] = load_frag(&As[wm * 32 + mt * 16 + ln][0], k0);
#pragma unroll
    for (int nt = 0; nt < 4; ++nt) {
      v16bf bfr = load_frag(&Bs[wn * 64 + nt * 16 + ln][0], k0);
#pragma unroll
      for (int mt = 0; mt < 2; ++mt)
        acc[mt][nt] = WMMA_BF16(af[mt], bfr, acc[mt][nt]);
    }
  }

  // --- epilogue: bias, scale, head-split store [B,H,T,DK] bf16 ---
#pragma unroll
  for (int mt = 0; mt < 2; ++mt)
#pragma unroll
    for (int nt = 0; nt < 4; ++nt) {
      const int gn = n0 + wn * 64 + nt * 16 + ln;
      const float bv_ = bias[gn];
      const int h  = gn >> 6, dk = gn & 63;
#pragma unroll
      for (int r = 0; r < 8; ++r) {
        const int gm = m0 + wm * 32 + mt * 16 + r + 8 * half;
        const int bb = gm >> 11, t = gm & 2047;
        dst[(((size_t)(bb * H + h)) * T + t) * DK + dk] =
            f2bf((acc[mt][nt][r] + bv_) * scale);
      }
    }
}

// ---------------------------------------------------------------------------
// Kernel 2: flash attention. One (b,h,q-block of 64) per 128-thread block.
// K tile staged by the Tensor Data Mover (with LDS row padding); V tile
// transposed manually. Each wave owns 16 q-rows.
// ---------------------------------------------------------------------------
__global__ __launch_bounds__(128) void attn(
    const unsigned short* __restrict__ Qu, const unsigned short* __restrict__ Ku,
    const unsigned short* __restrict__ Vu, unsigned short* __restrict__ Yu) {
  const __bf16* Q = (const __bf16*)Qu;
  const __bf16* K = (const __bf16*)Ku;
  const __bf16* V = (const __bf16*)Vu;
  __bf16* Y = (__bf16*)Yu;

  const int QB = T / 64;
  const int bh = blockIdx.x / QB, qb = blockIdx.x % QB;
  const int b = bh >> 4, h = bh & 15;
  const int tid = threadIdx.x, wave = tid >> 5;
  const int lane = tid & 31, ln = lane & 15, half = lane >> 4, k0 = half * 8;

  __shared__ __align__(16) __bf16 Ks[32][72];      // [key][dk]; 128B data+16B pad
  __shared__ __align__(16) __bf16 Vs[64][48];      // [dk][key] (transposed)
  __shared__ __align__(16) __bf16 Ps[4][16][48];   // per-wave P re-layout

  const unsigned int ks_off = (unsigned int)(size_t)(void*)&Ks[0][0];

  // Q fragments: persistent in registers (16 rows x 64 dk = 2 K-chunks)
  const __bf16* qbase = Q + ((size_t)bh * T + qb * 64 + wave * 16 + ln) * DK;
  const v16bf qf0 = load_frag(qbase, k0);        // d = 0..31
  const v16bf qf1 = load_frag(qbase + 32, k0);   // d = 32..63

  v8f O[4];
#pragma unroll
  for (int t = 0; t < 4; ++t) O[t] = zero8();
  float mi[8], li[8];
#pragma unroll
  for (int r = 0; r < 8; ++r) { mi[r] = -1e30f; li[r] = 0.0f; }

  const int nkb = qb * 2 + 2;            // keys [0, qb*64+64) in blocks of 32
  const int srow = tid >> 2, scb = (tid & 3) * 16;

  for (int kb = 0; kb < nkb; ++kb) {
    const int kk = kb * 32;
    __syncthreads();
    // --- K tile via TDM: 32 rows x 64 bf16, LDS rows padded 128B+16B ---
    if (wave == 0) {
      tdm_load_2d_bf16(ks_off, K + ((size_t)bh * T + kk) * DK,
                       /*tile_w=*/DK, /*tile_h=*/32, /*row_stride=*/DK,
                       /*pad_interval(32dw)=*/4, /*pad_amount(4dw)=*/3);
    }
    // --- V tile staged transposed by all threads ---
    {
      const v8bf* vp = (const v8bf*)(V + ((size_t)bh * T + kk + srow) * DK + scb);
      v8bf v0 = vp[0], v1 = vp[1];
#pragma unroll
      for (int j = 0; j < 8; ++j) {
        Vs[scb + j][srow]     = v0[j];
        Vs[scb + 8 + j][srow] = v1[j];
      }
    }
    if (wave == 0) __builtin_amdgcn_s_wait_tensorcnt(0);
    __syncthreads();

    // --- S = Q K^T : two 16x16 N-tiles, DK split into two K=32 chunks ---
    v8f s0 = zero8(), s1 = zero8();
    {
      v16bf kf;
      kf = load_frag(&Ks[ln][0],       k0); s0 = WMMA_BF16(qf0, kf, s0);
      kf = load_frag(&Ks[ln][32],      k0); s0 = WMMA_BF16(qf1, kf, s0);
      kf = load_frag(&Ks[16 + ln][0],  k0); s1 = WMMA_BF16(qf0, kf, s1);
      kf = load_frag(&Ks[16 + ln][32], k0); s1 = WMMA_BF16(qf1, kf, s1);
    }

    // --- causal mask + online softmax (rows r+8*half, cols = lanes) ---
    const int qrow0 = qb * 64 + wave * 16 + 8 * half;
    const int j0 = kk + ln, j1 = kk + 16 + ln;
    float pr0[8], pr1[8];
#pragma unroll
    for (int r = 0; r < 8; ++r) {
      const int q = qrow0 + r;
      float a0 = (j0 <= q) ? s0[r] : -1e30f;
      float a1 = (j1 <= q) ? s1[r] : -1e30f;
      float vmax = fmaxf(a0, a1);
      vmax = fmaxf(vmax, __shfl_xor(vmax, 1, 32));
      vmax = fmaxf(vmax, __shfl_xor(vmax, 2, 32));
      vmax = fmaxf(vmax, __shfl_xor(vmax, 4, 32));
      vmax = fmaxf(vmax, __shfl_xor(vmax, 8, 32));
      const float mn = fmaxf(mi[r], vmax);
      const float alpha = exp2f((mi[r] - mn) * 1.44269504f);
      const float p0 = exp2f((a0 - mn) * 1.44269504f);
      const float p1 = exp2f((a1 - mn) * 1.44269504f);
      float ps = p0 + p1;
      ps += __shfl_xor(ps, 1, 32);
      ps += __shfl_xor(ps, 2, 32);
      ps += __shfl_xor(ps, 4, 32);
      ps += __shfl_xor(ps, 8, 32);
      li[r] = li[r] * alpha + ps;
      mi[r] = mn;
      O[0][r] *= alpha; O[1][r] *= alpha; O[2][r] *= alpha; O[3][r] *= alpha;
      pr0[r] = p0; pr1[r] = p1;
    }

    // --- P: C-layout -> A-frag layout via per-wave LDS (same-wave DS in-order)
    __bf16 (*ps)[48] = Ps[wave];
#pragma unroll
    for (int r = 0; r < 8; ++r) {
      ps[r + 8 * half][ln]      = f2bf(pr0[r]);
      ps[r + 8 * half][16 + ln] = f2bf(pr1[r]);
    }
    const v16bf pf = load_frag(&ps[ln][0], k0);

    // --- O += P V : 4 dk N-tiles, K=32 keys ---
    {
      v16bf vf;
      vf = load_frag(&Vs[ln][0],      k0); O[0] = WMMA_BF16(pf, vf, O[0]);
      vf = load_frag(&Vs[16 + ln][0], k0); O[1] = WMMA_BF16(pf, vf, O[1]);
      vf = load_frag(&Vs[32 + ln][0], k0); O[2] = WMMA_BF16(pf, vf, O[2]);
      vf = load_frag(&Vs[48 + ln][0], k0); O[3] = WMMA_BF16(pf, vf, O[3]);
    }
  }

  // --- epilogue: normalize, store bf16 [B,T,D] (merge heads) ---
#pragma unroll
  for (int t = 0; t < 4; ++t) {
    const int col = h * DK + t * 16 + ln;
#pragma unroll
    for (int r = 0; r < 8; ++r) {
      const int qrow = qb * 64 + wave * 16 + r + 8 * half;
      Y[((size_t)b * T + qrow) * D + col] = f2bf(O[t][r] / li[r]);
    }
  }
}

// ---------------------------------------------------------------------------
// Kernel 3: output projection, out = y @ Wo^T + bo (bf16 A via TDM, f32 out).
// ---------------------------------------------------------------------------
__global__ __launch_bounds__(256) void out_gemm(
    const unsigned short* __restrict__ Yu, const float* __restrict__ Wo,
    const float* __restrict__ bo, float* __restrict__ out) {
  const __bf16* Y = (const __bf16*)Yu;
  const int m0 = blockIdx.x * 128, n0 = blockIdx.y * 128;
  const int tid  = threadIdx.x;
  const int lane = tid & 31, ln = lane & 15, half = lane >> 4, k0 = half * 8;
  const int wave = tid >> 5, wm = wave >> 1, wn = wave & 1;

  __shared__ __align__(16) __bf16 As[128][LDS_STRIDE];  // 64B data + 32B pad
  __shared__ __align__(16) __bf16 Bs[128][LDS_STRIDE];

  const unsigned int as_off = (unsigned int)(size_t)(void*)&As[0][0];

  v8f acc[2][4];
#pragma unroll
  for (int i = 0; i < 2; ++i)
#pragma unroll
    for (int j = 0; j < 4; ++j) acc[i][j] = zero8();

  const int srow = tid >> 1, scb = (tid & 1) * 16;

  for (int kk = 0; kk < D; kk += 32) {
    __syncthreads();
    // --- A tile via TDM: 128 rows x 32 bf16, LDS rows padded 64B+32B ---
    if (wave == 0) {
      tdm_load_2d_bf16(as_off, Y + (size_t)m0 * D + kk,
                       /*tile_w=*/32, /*tile_h=*/128, /*row_stride=*/D,
                       /*pad_interval(16dw)=*/3, /*pad_amount(8dw)=*/7);
    }
    // --- B tile (Wo, f32 -> bf16) staged by all threads ---
    {
      const float4* bp = (const float4*)(Wo + (size_t)(n0 + srow) * D + kk + scb);
      if (kk + 32 < D)
        __builtin_prefetch(Wo + (size_t)(n0 + srow) * D + kk + 32 + scb, 0, 1);
      float4 b0 = bp[0], b1 = bp[1], b2 = bp[2], b3 = bp[3];
      v8bf pb0, pb1;
      pb0[0]=f2bf(b0.x); pb0[1]=f2bf(b0.y); pb0[2]=f2bf(b0.z); pb0[3]=f2bf(b0.w);
      pb0[4]=f2bf(b1.x); pb0[5]=f2bf(b1.y); pb0[6]=f2bf(b1.z); pb0[7]=f2bf(b1.w);
      pb1[0]=f2bf(b2.x); pb1[1]=f2bf(b2.y); pb1[2]=f2bf(b2.z); pb1[3]=f2bf(b2.w);
      pb1[4]=f2bf(b3.x); pb1[5]=f2bf(b3.y); pb1[6]=f2bf(b3.z); pb1[7]=f2bf(b3.w);
      *(v8bf*)&Bs[srow][scb]     = pb0;
      *(v8bf*)&Bs[srow][scb + 8] = pb1;
    }
    if (wave == 0) __builtin_amdgcn_s_wait_tensorcnt(0);
    __syncthreads();

    v16bf af[2];
#pragma unroll
    for (int mt = 0; mt < 2; ++mt)
      af[mt] = load_frag(&As[wm * 32 + mt * 16 + ln][0], k0);
#pragma unroll
    for (int nt = 0; nt < 4; ++nt) {
      v16bf bfr = load_frag(&Bs[wn * 64 + nt * 16 + ln][0], k0);
#pragma unroll
      for (int mt = 0; mt < 2; ++mt)
        acc[mt][nt] = WMMA_BF16(af[mt], bfr, acc[mt][nt]);
    }
  }

#pragma unroll
  for (int mt = 0; mt < 2; ++mt)
#pragma unroll
    for (int nt = 0; nt < 4; ++nt) {
      const int gn = n0 + wn * 64 + nt * 16 + ln;
      const float bv_ = bo[gn];
#pragma unroll
      for (int r = 0; r < 8; ++r) {
        const int gm = m0 + wm * 32 + mt * 16 + r + 8 * half;
        out[(size_t)gm * D + gn] = acc[mt][nt][r] + bv_;
      }
    }
}

// ---------------------------------------------------------------------------
extern "C" void kernel_launch(void* const* d_in, const int* in_sizes, int n_in,
                              void* d_out, int out_size, void* d_ws, size_t ws_size,
                              hipStream_t stream) {
  (void)in_sizes; (void)n_in; (void)out_size; (void)ws_size;
  const float* x  = (const float*)d_in[0];
  // d_in[1] = mask (causal tril) — handled analytically in attn kernel
  const float* Wq = (const float*)d_in[2];
  const float* bq = (const float*)d_in[3];
  const float* Wk = (const float*)d_in[4];
  const float* bk = (const float*)d_in[5];
  const float* Wv = (const float*)d_in[6];
  const float* bv = (const float*)d_in[7];
  const float* Wo = (const float*)d_in[8];
  const float* bo = (const float*)d_in[9];

  const size_t elems = (size_t)Bn * H * T * DK;   // 8,388,608 bf16 elements
  unsigned short* Qu = (unsigned short*)d_ws;
  unsigned short* Ku = Qu + elems;
  unsigned short* Vu = Ku + elems;
  unsigned short* Yu = Vu + elems;                // [B,T,D] bf16

  dim3 g1(M_TOT / 128, D / 128, 3);               // 64 x 8 x 3
  qkv_gemm<<<g1, 256, 0, stream>>>(x, Wq, bq, Wk, bk, Wv, bv, Qu, Ku, Vu);

  dim3 g2((Bn * H) * (T / 64));                   // 2048 blocks
  attn<<<g2, 128, 0, stream>>>(Qu, Ku, Vu, Yu);

  dim3 g3(M_TOT / 128, D / 128, 1);
  out_gemm<<<g3, 256, 0, stream>>>(Yu, Wo, bo, (float*)d_out);
}
